// MultiHeadAttentionLayer_65420941853357
// MI455X (gfx1250) — matrix operand order repaired and checked by reading, not compiled, not run on hardware
//
#include <hip/hip_runtime.h>
#include <hip/hip_bf16.h>
#include <math.h>

// Graph-transformer MHA layer for MI455X (gfx1250, wave32).
//   N=40000 nodes, E=640000 edges, IN=128, H=8 heads, D=16 -> H*D=128.
//
// All GEMMs run as split-fp16 3-product WMMA (fp32 ~ hi+lo f16 halves):
//   A*B ~= Ah*Bh + Ah*Bl + Al*Bh   with V_WMMA_F32_16X16X32_F16, fp32 accum.
// ~2^-21 relative error (near fp32) at the f16 matrix rate, so the fused
// edge kernel is bound by the unavoidable e/e_out HBM traffic (~660 MB).
//
// Pipeline:
//   0) swz_kernel : swizzle Wq/Wk/Wv/We into per-lane B-fragment order (f16
//                   hi/lo), 256 KB in workspace, L2-resident.
//   1) qkv_kernel : stage h tile once (f16 hi/lo in LDS), compute Q,K,V.
//   2) edge_kernel: fused pe = e @ We + be (WMMA accumulators, never hits HBM)
//                   -> score = K[src]*Q[dst]/4 * pe -> e_out
//                   -> att = exp(clamp(rowsum))     (shfl_xor reduce)
//                   -> atomic scatter V[src]*att -> h_out, att -> z.
//   3) norm_kernel: h_out /= (z + 1e-6).

typedef _Float16 v4h  __attribute__((ext_vector_type(4)));
typedef _Float16 v8h  __attribute__((ext_vector_type(8)));
typedef _Float16 v16h __attribute__((ext_vector_type(16)));
typedef float    v8f  __attribute__((ext_vector_type(8)));

#define F32_STRIDE 132   // f32 LDS row stride (128+4): conflict-free
#define F16_STRIDE 136   // f16 LDS row stride in halves (16B-aligned rows)

// ---------------------------------------------------------------- helpers
__device__ __forceinline__ void split_store4(float4 v,
                                             _Float16* __restrict__ hi,
                                             _Float16* __restrict__ lo) {
    v4h h, l;
    h[0] = (_Float16)v.x;  l[0] = (_Float16)(v.x - (float)h[0]);
    h[1] = (_Float16)v.y;  l[1] = (_Float16)(v.y - (float)h[1]);
    h[2] = (_Float16)v.z;  l[2] = (_Float16)(v.z - (float)h[2]);
    h[3] = (_Float16)v.w;  l[3] = (_Float16)(v.w - (float)h[3]);
    *(v4h*)hi = h;
    *(v4h*)lo = l;
}

// One 16x16 (M x N) x K=128 tile: split-f16 3-product accumulation.
// aHi/aLo: LDS, 16 rows x F16_STRIDE halves. FH/FL: per-(mat,head) fragment
// base (layout: [kc:4][lane:32][16 halves], built by swz_kernel).
__device__ __forceinline__ v8f gemm_tile_f16x3(
    const _Float16* __restrict__ aHi, const _Float16* __restrict__ aLo,
    const _Float16* __restrict__ FH,  const _Float16* __restrict__ FL,
    int lane)
{
    const int half = lane >> 4;
    const int lc   = lane & 15;
    v8f acc = {};
    #pragma unroll
    for (int kc = 0; kc < 4; ++kc) {
        const int kb = kc * 32;
        const int o  = lc * F16_STRIDE + kb + half * 8;
        // A fragment: two contiguous 8-half runs per ISA 16-bit A layout.
        v8h a0h = *(const v8h*)&aHi[o];
        v8h a1h = *(const v8h*)&aHi[o + 16];
        v8h a0l = *(const v8h*)&aLo[o];
        v8h a1l = *(const v8h*)&aLo[o + 16];
        v16h Ah = __builtin_shufflevector(a0h, a1h,
                    0,1,2,3,4,5,6,7,8,9,10,11,12,13,14,15);
        v16h Al = __builtin_shufflevector(a0l, a1l,
                    0,1,2,3,4,5,6,7,8,9,10,11,12,13,14,15);
        // B fragment: pre-swizzled, one coalesced 32B load per operand.
        v16h Bh = *(const v16h*)&FH[(kc * 32 + lane) * 16];
        v16h Bl = *(const v16h*)&FL[(kc * 32 + lane) * 16];
        acc = __builtin_amdgcn_wmma_f32_16x16x32_f16(
                  false, Ah, false, Bh, (short)0, acc, false, false);
        acc = __builtin_amdgcn_wmma_f32_16x16x32_f16(
                  false, Ah, false, Bl, (short)0, acc, false, false);
        acc = __builtin_amdgcn_wmma_f32_16x16x32_f16(
                  false, Al, false, Bh, (short)0, acc, false, false);
    }
    return acc;
}

// ---------------------------------------------------------------- zero fill
__global__ __launch_bounds__(256) void zero_kernel(float* __restrict__ p, int n) {
    int i = blockIdx.x * 256 + threadIdx.x;
    if (i < n) p[i] = 0.0f;
}

// -------------------------------------------------- weight fragment swizzle
// grid = 4 mats * 8 heads * 4 kc = 128 blocks, 256 threads.
// Fragment element (lane,i) of (mat,w,kc) = W[kc*32 + 8*(lane>>4) + i + 8*(i>=8)]
//                                            [w*16 + (lane&15)]  (f16 hi/lo).
__global__ __launch_bounds__(256) void swz_kernel(
    const float* __restrict__ Wq, const float* __restrict__ Wk,
    const float* __restrict__ Wv, const float* __restrict__ We,
    _Float16* __restrict__ FH, _Float16* __restrict__ FL)
{
    const int bx  = blockIdx.x;
    const int mat = bx >> 5;
    const int rem = bx & 31;
    const int w   = rem >> 2;
    const int kc  = rem & 3;
    const float* W = (mat == 0) ? Wq : (mat == 1) ? Wk : (mat == 2) ? Wv : We;
    const size_t base = (((size_t)(mat * 8 + w) * 4) + kc) * 512;

    #pragma unroll
    for (int elem = threadIdx.x; elem < 512; elem += 256) {
        int lane = elem >> 4;
        int i    = elem & 15;
        int half = lane >> 4;
        int lcc  = lane & 15;
        int col  = w * 16 + lcc;
        int k    = kc * 32 + 8 * half + i + ((i >= 8) ? 8 : 0);
        float x  = W[(size_t)k * 128 + col];
        _Float16 hi = (_Float16)x;
        _Float16 lo = (_Float16)(x - (float)hi);
        FH[base + elem] = hi;
        FL[base + elem] = lo;
    }
}

// ------------------------------------------------------- Q/K/V projections
// grid = n/16 blocks, 256 threads (8 waves). Wave w = head w; each block
// stages the h tile ONCE and produces Q, K and V from the same A fragments.
__global__ __launch_bounds__(256) void qkv_kernel(
    const float* __restrict__ hG,
    const float* __restrict__ bq, const float* __restrict__ bk,
    const float* __restrict__ bv,
    const _Float16* __restrict__ FH, const _Float16* __restrict__ FL,
    float* __restrict__ qkvOut,   // [3][n][128]
    int nNodes)
{
    __shared__ __align__(16) _Float16 aHi[16 * F16_STRIDE];
    __shared__ __align__(16) _Float16 aLo[16 * F16_STRIDE];

    const int tid   = threadIdx.x;
    const int node0 = blockIdx.x * 16;

    #pragma unroll
    for (int i = tid; i < 512; i += 256) {
        int row = i >> 5;
        int c4  = (i & 31) << 2;
        float4 v = *(const float4*)&hG[(size_t)(node0 + row) * 128 + c4];
        split_store4(v, &aHi[row * F16_STRIDE + c4], &aLo[row * F16_STRIDE + c4]);
    }
    __syncthreads();

    const int lane = tid & 31;
    const int wv   = tid >> 5;
    const int half = lane >> 4;
    const int lc   = lane & 15;
    const int col  = wv * 16 + lc;

    const float* biases[3] = { bq, bk, bv };

    #pragma unroll
    for (int sel = 0; sel < 3; ++sel) {
        const size_t fb = (size_t)(sel * 8 + wv) * 2048;   // 4*32*16 halves
        v8f acc = gemm_tile_f16x3(aHi, aLo, FH + fb, FL + fb, lane);
        const float bias = biases[sel][col];
        float* outB = qkvOut + (size_t)sel * (size_t)nNodes * 128;
        #pragma unroll
        for (int r = 0; r < 8; ++r) {
            int m = r + half * 8;
            outB[(size_t)(node0 + m) * 128 + col] = acc[r] + bias;
        }
    }
}

// ------------------------------------------------------ fused edge kernel
// grid = E/16 blocks, 256 threads (8 waves, wave w = head w).
__global__ __launch_bounds__(256) void edge_kernel(
    const float* __restrict__ eG,
    const int*   __restrict__ srcG, const int* __restrict__ dstG,
    const float* __restrict__ beG,
    const _Float16* __restrict__ FH, const _Float16* __restrict__ FL,
    const float* __restrict__ Qv,   // [n][128]
    const float* __restrict__ Kv,
    const float* __restrict__ Vv,
    float* __restrict__ eOut,       // [E][128]
    float* __restrict__ hAcc,       // [n][128]  (atomic accumulate)
    float* __restrict__ zAcc)       // [n][8]    (atomic accumulate)
{
    __shared__ __align__(16) _Float16 aHi[16 * F16_STRIDE];
    __shared__ __align__(16) _Float16 aLo[16 * F16_STRIDE];
    __shared__ float kLds[16 * F32_STRIDE];
    __shared__ float qLds[16 * F32_STRIDE];
    __shared__ float vLds[16 * F32_STRIDE];
    __shared__ int   sIdx[16];
    __shared__ int   dIdx[16];

    const int tid = threadIdx.x;
    const int e0  = blockIdx.x * 16;

    if (tid < 16) {
        sIdx[tid] = srcG[e0 + tid];
        dIdx[tid] = dstG[e0 + tid];
    }
    __syncthreads();

    // Stage streaming e rows (split to f16 hi/lo for WMMA A) and the gathered
    // K[src]/Q[dst]/V[src] rows (node arrays are L2-resident: 20MB vs 192MB).
    #pragma unroll
    for (int i = tid; i < 512; i += 256) {
        int row = i >> 5;
        int c4  = (i & 31) << 2;
        int sn  = sIdx[row];
        int dn  = dIdx[row];
        float4 ev = *(const float4*)&eG[(size_t)(e0 + row) * 128 + c4];
        split_store4(ev, &aHi[row * F16_STRIDE + c4], &aLo[row * F16_STRIDE + c4]);
        *(float4*)&kLds[row * F32_STRIDE + c4] =
            *(const float4*)&Kv[(size_t)sn * 128 + c4];
        *(float4*)&qLds[row * F32_STRIDE + c4] =
            *(const float4*)&Qv[(size_t)dn * 128 + c4];
        *(float4*)&vLds[row * F32_STRIDE + c4] =
            *(const float4*)&Vv[(size_t)sn * 128 + c4];
    }
    __syncthreads();

    const int lane = tid & 31;
    const int wv   = tid >> 5;        // head id
    const int half = lane >> 4;
    const int lc   = lane & 15;
    const int col  = wv * 16 + lc;    // column in [0,128): head*16 + d

    // pe tile = e_tile @ We (We fragments are mat index 3).
    const size_t fb = (size_t)(3 * 8 + wv) * 2048;
    v8f acc = gemm_tile_f16x3(aHi, aLo, FH + fb, FL + fb, lane);

    const float bias = beG[col];
    const float inv_scale = 0.25f;    // 1/sqrt(D), D=16

    #pragma unroll
    for (int r = 0; r < 8; ++r) {
        const int m = r + half * 8;   // edge slot within tile (C/D layout)
        float pe = acc[r] + bias;
        float kq = kLds[m * F32_STRIDE + col] * qLds[m * F32_STRIDE + col];
        float score = kq * inv_scale * pe;

        // e_out ('score' edge features)
        eOut[(size_t)(e0 + m) * 128 + col] = score;

        // row-sum over d: 16 lanes of each half-wave hold one edge row.
        float s = score;
        s += __shfl_xor(s, 8);
        s += __shfl_xor(s, 4);
        s += __shfl_xor(s, 2);
        s += __shfl_xor(s, 1);
        float att = __expf(fminf(fmaxf(s, -5.0f), 5.0f));

        const int dn = dIdx[m];
        float val = vLds[m * F32_STRIDE + col] * att;
        atomicAdd(&hAcc[(size_t)dn * 128 + col], val);
        if (lc == 0) atomicAdd(&zAcc[(size_t)dn * 8 + wv], att);
    }
}

// ------------------------------------------------------------ h_out /= z
__global__ __launch_bounds__(256) void norm_kernel(
    float* __restrict__ hOut, const float* __restrict__ zAcc, int total)
{
    int i = blockIdx.x * 256 + threadIdx.x;
    if (i >= total) return;
    int n = i >> 7;
    int c = i & 127;
    int hd = c >> 4;
    hOut[i] = hOut[i] / (zAcc[n * 8 + hd] + 1e-6f);
}

// ------------------------------------------------------------------ launch
extern "C" void kernel_launch(void* const* d_in, const int* in_sizes, int n_in,
                              void* d_out, int out_size, void* d_ws, size_t ws_size,
                              hipStream_t stream) {
    const float* h   = (const float*)d_in[0];
    const float* e   = (const float*)d_in[1];
    const int*   src = (const int*)  d_in[2];
    const int*   dst = (const int*)  d_in[3];
    const float* Wq  = (const float*)d_in[4];
    const float* bq  = (const float*)d_in[5];
    const float* Wk  = (const float*)d_in[6];
    const float* bk  = (const float*)d_in[7];
    const float* Wv  = (const float*)d_in[8];
    const float* bv  = (const float*)d_in[9];
    const float* We  = (const float*)d_in[10];
    const float* be  = (const float*)d_in[11];

    const int n  = in_sizes[0] / 128;   // 40000
    const int ne = in_sizes[2];         // 640000

    float* out  = (float*)d_out;
    float* hOut = out;                            // [n][128]
    float* eOut = out + (size_t)n * 128;          // [E][128]

    float* ws   = (float*)d_ws;
    float* Qv   = ws;                              // [n][128]
    float* Kv   = ws + (size_t)n * 128;
    float* Vv   = ws + (size_t)2 * n * 128;
    float* zAcc = ws + (size_t)3 * n * 128;        // [n][8]
    // Weight fragments (f16): 4 mats * 8 heads * 4 kc * 32 lanes * 16 halves.
    _Float16* FH = (_Float16*)(ws + (size_t)3 * n * 128 + (size_t)n * 8);
    _Float16* FL = FH + (size_t)4 * 8 * 4 * 512;   // 65536 halves each

    // Zero accumulators (graph-capture safe, deterministic each replay).
    {
        int t1 = n * 128;
        zero_kernel<<<(t1 + 255) / 256, 256, 0, stream>>>(hOut, t1);
        int t2 = n * 8;
        zero_kernel<<<(t2 + 255) / 256, 256, 0, stream>>>(zAcc, t2);
    }

    // Swizzle weights into WMMA B-fragment order (f16 hi/lo).
    swz_kernel<<<128, 256, 0, stream>>>(Wq, Wk, Wv, We, FH, FL);

    // Q/K/V projections (h staged once per tile).
    qkv_kernel<<<n / 16, 256, 0, stream>>>(h, bq, bk, bv, FH, FL, ws, n);

    // Fused edge GEMM + score + softmax-numerator + scatter.
    edge_kernel<<<ne / 16, 256, 0, stream>>>(e, src, dst, be, FH, FL,
                                             Qv, Kv, Vv, eOut, hOut, zAcc);

    // Normalize.
    {
        int t = n * 128;
        norm_kernel<<<(t + 255) / 256, 256, 0, stream>>>(hOut, zAcc, t);
    }
}
